// TreeGNN_15960098472357
// MI455X (gfx1250) — matrix operand order repaired and checked by reading, not compiled, not run on hardware
//
#include <hip/hip_runtime.h>
#include <hip/hip_bf16.h>
#include <stdint.h>

typedef __attribute__((ext_vector_type(16))) _Float16 v16h;
typedef __attribute__((ext_vector_type(8)))  float    v8f;

#define N_NODES 50000
#define N_EDGES 600000
#define NG      512
#define HD      128
#define NC      3
#define BN_EPS  1e-5f
#define NTILES  (N_NODES / 16)   // 3125, exact

// workspace layout (float offsets, generously padded)
#define OFF_DINV  0ll          // N
#define OFF_H     50048ll      // N*129 (layer1) / N*128 later
#define OFF_HW    6500096ll    // N*128
#define OFF_AGG   12900160ll   // N*128
#define OFF_BN    19300224ll   // 256 (sum, sumsq)
#define OFF_POOL  19300480ll   // NG*128
#define OFF_CNT   (OFF_POOL + (long long)NG*HD)          // NG
#define OFF_HPK   19366784ll   // packed A fragments: up to 8M halfs (4M floats)
#define OFF_WPK   23366784ll   // packed B fragments: up to 20480 halfs

// ---------------- utility kernels ----------------
__global__ void k_fill(float* p, float v, int n) {
  int i = blockIdx.x * blockDim.x + threadIdx.x;
  if (i < n) p[i] = v;
}

__global__ void k_deg_scatter(const int* __restrict__ dst, float* __restrict__ deg) {
  int i = blockIdx.x * blockDim.x + threadIdx.x;
  if (i < N_EDGES) atomicAdd(&deg[dst[i]], 1.0f);
}

__global__ void k_rsqrt_inplace(float* p, int n) {
  int i = blockIdx.x * blockDim.x + threadIdx.x;
  if (i < n) p[i] = rsqrtf(p[i]);
}

// build h0 = [pos | emb0[i0] | emb1[i1] | emb2[i2] | emb5[i5]]  (N x 129)
__global__ void k_embed(const float* __restrict__ x,
                        const float* __restrict__ e0, const float* __restrict__ e1,
                        const float* __restrict__ e2, const float* __restrict__ e5,
                        float* __restrict__ h) {
  int i = blockIdx.x * blockDim.x + threadIdx.x;
  if (i >= N_NODES * 129) return;
  int n = i / 129, f = i % 129;
  float v;
  if (f == 0)        v = x[n * 5 + 0];
  else if (f < 33)   v = e0[(int)x[n * 5 + 1] * 32 + (f - 1)];
  else if (f < 65)   v = e1[(int)x[n * 5 + 2] * 32 + (f - 33)];
  else if (f < 97)   v = e2[(int)x[n * 5 + 3] * 32 + (f - 65)];
  else               v = e5[(int)x[n * 5 + 4] * 32 + (f - 97)];
  h[i] = v;
}

// ---------------- fragment packing (ISA 7.12.2 layouts) ----------------
// A: per (row-tile m, k-chunk kc, lane): 16 halfs contiguous.
// element e of lane: v=e>>1, kk=e&1; K = kc*32 + (v>>2)*16 + (lane>>4)*8 + (v&3)*2 + kk
__global__ void k_packA(const float* __restrict__ h, _Float16* __restrict__ apk,
                        int K, int KC, int total) {
  int i = blockIdx.x * blockDim.x + threadIdx.x;
  if (i >= total) return;
  int e    = i & 15;
  int lane = (i >> 4) & 31;
  int kc   = (i >> 9) % KC;
  int m    = i / (512 * KC);
  int v = e >> 1, kk = e & 1;
  int hi = lane >> 4, lrow = lane & 15;
  int row = m * 16 + lrow;
  int k = kc * 32 + ((v >> 2) << 4) + (hi << 3) + ((v & 3) << 1) + kk;
  float val = (k < K) ? h[row * K + k] : 0.0f;
  apk[i] = (_Float16)val;
}

// B: per (k-chunk kc, n-tile t, lane): 16 halfs contiguous.
// element e of lane: j=e>>1, kk=e&1; K = kc*32 + (j>>2)*16 + (lane>>4)*8 + (j&3)*2 + kk; N = t*16 + (lane&15)
__global__ void k_packB(const float* __restrict__ W, _Float16* __restrict__ bpk,
                        int K, int KC, int total) {
  int i = blockIdx.x * blockDim.x + threadIdx.x;
  if (i >= total) return;
  int e    = i & 15;
  int lane = (i >> 4) & 31;
  int t    = (i >> 9) & 7;
  int kc   = i >> 12;
  int j = e >> 1, kk = e & 1;
  int hi = lane >> 4;
  int col = (t << 4) + (lane & 15);
  int k = kc * 32 + ((j >> 2) << 4) + (hi << 3) + ((j & 3) << 1) + kk;
  float val = (k < K) ? W[k * HD + col] : 0.0f;
  bpk[i] = (_Float16)val;
}

// ---------------- WMMA GEMM: out[N x 128] = h[N x K] @ W[K x 128] ----------------
// one wave per 16-row tile; A fragments from packed global (b128 loads),
// B fragments from LDS-staged packed weights (ds_load_b128).
template <int KC>
__global__ void k_gemm_wmma(const _Float16* __restrict__ apk,
                            const _Float16* __restrict__ bpk,
                            float* __restrict__ out) {
  __shared__ __align__(32) _Float16 ldsW[KC * 8 * 512];
  // cooperative copy of packed W into LDS (KC*8*256 dwords)
  {
    const uint32_t* src = (const uint32_t*)bpk;
    uint32_t*       dstp = (uint32_t*)ldsW;
    for (int idx = threadIdx.x; idx < KC * 8 * 256; idx += blockDim.x)
      dstp[idx] = src[idx];
  }
  __syncthreads();

  int wave = blockIdx.x * (blockDim.x >> 5) + (threadIdx.x >> 5);
  int lane = threadIdx.x & 31;
  if (wave >= NTILES) return;            // wave-uniform: EXEC all-ones at WMMA
  int lrow = lane & 15;
  int hi   = lane >> 4;

  const v16h* ap = (const v16h*)apk + (size_t)wave * KC * 32 + lane;
  const v16h* bp = (const v16h*)ldsW + lane;

  v8f acc[8];
  v8f zero = {};
#pragma unroll
  for (int t = 0; t < 8; ++t) acc[t] = zero;

#pragma unroll
  for (int kc = 0; kc < KC; ++kc) {
    v16h a = ap[kc * 32];
#pragma unroll
    for (int t = 0; t < 8; ++t) {
      v16h b = bp[(kc * 8 + t) * 32];
      acc[t] = __builtin_amdgcn_wmma_f32_16x16x32_f16(
          /*neg_a=*/false, a, /*neg_b=*/false, b,
          /*c_mod=*/(short)0, acc[t], /*reuse_a=*/false, /*reuse_b=*/false);
    }
  }
  // D layout: vgpr r -> M = r + 8*hi, N = lane&15
  int row0 = wave << 4;
#pragma unroll
  for (int t = 0; t < 8; ++t) {
#pragma unroll
    for (int r = 0; r < 8; ++r) {
      int rr = row0 + r + (hi << 3);
      out[rr * HD + (t << 4) + lrow] = acc[t][r];
    }
  }
}

// ---------------- aggregation ----------------
__global__ void k_selfinit(const float* __restrict__ hw, const float* __restrict__ dinv,
                           const float* __restrict__ b, float* __restrict__ agg) {
  int i = blockIdx.x * blockDim.x + threadIdx.x;
  if (i >= N_NODES * HD) return;
  int n = i / HD, c = i % HD;
  float di = dinv[n];
  agg[i] = hw[i] * di * di + b[c];
}

__global__ void k_edge_scatter(const int* __restrict__ src, const int* __restrict__ dst,
                               const float* __restrict__ dinv, const float* __restrict__ hw,
                               float* __restrict__ agg) {
  long long t = (long long)blockIdx.x * blockDim.x + threadIdx.x;
  if (t >= (long long)N_EDGES * 32) return;
  int e = (int)(t >> 5);
  int q = (int)(t & 31);
  int s = src[e], d = dst[e];
  float norm = dinv[s] * dinv[d];
  int c0 = q * 4;
#pragma unroll
  for (int k = 0; k < 4; ++k)
    atomicAdd(&agg[d * HD + c0 + k], hw[s * HD + c0 + k] * norm);
}

// ---------------- batch norm ----------------
__global__ void k_bnstats(const float* __restrict__ a, float* __restrict__ sums) {
  int c = threadIdx.x;                      // 128 threads
  float s = 0.f, s2 = 0.f;
  for (int n = blockIdx.x; n < N_NODES; n += gridDim.x) {
    float v = a[n * HD + c];
    s += v; s2 += v * v;
  }
  atomicAdd(&sums[c], s);
  atomicAdd(&sums[HD + c], s2);
}

__global__ void k_bnapply_relu(const float* __restrict__ a, const float* __restrict__ sums,
                               const float* __restrict__ g, const float* __restrict__ be,
                               float* __restrict__ hnext) {
  int i = blockIdx.x * blockDim.x + threadIdx.x;
  if (i >= N_NODES * HD) return;
  int c = i % HD;
  const float invN = 1.0f / (float)N_NODES;
  float mean = sums[c] * invN;
  float var  = sums[HD + c] * invN - mean * mean;
  float v = (a[i] - mean) * rsqrtf(var + BN_EPS) * g[c] + be[c];
  hnext[i] = v > 0.f ? v : 0.f;
}

// ---------------- pooling + fc ----------------
__global__ void k_pool_scatter(const float* __restrict__ h, const int* __restrict__ batch,
                               float* __restrict__ pool) {
  int i = blockIdx.x * blockDim.x + threadIdx.x;
  if (i >= N_NODES * HD) return;
  int n = i / HD, c = i % HD;
  atomicAdd(&pool[batch[n] * HD + c], h[i]);
}

__global__ void k_cnt_scatter(const int* __restrict__ batch, float* __restrict__ cnt) {
  int i = blockIdx.x * blockDim.x + threadIdx.x;
  if (i < N_NODES) atomicAdd(&cnt[batch[i]], 1.0f);
}

__global__ void k_final(const float* __restrict__ pool, const float* __restrict__ cnt,
                        const float* __restrict__ fcW, const float* __restrict__ fcb,
                        float* __restrict__ out) {
  int i = blockIdx.x * blockDim.x + threadIdx.x;
  if (i >= NG * NC) return;
  int g = i / NC, k = i % NC;
  float cn = cnt[g]; if (cn < 1.f) cn = 1.f;
  float inv = 1.0f / cn;
  float acc = fcb[k];
  for (int c = 0; c < HD; ++c)
    acc += pool[g * HD + c] * inv * fcW[c * NC + k];
  out[i] = acc;
}

// ---------------- launcher ----------------
extern "C" void kernel_launch(void* const* d_in, const int* in_sizes, int n_in,
                              void* d_out, int out_size, void* d_ws, size_t ws_size,
                              hipStream_t stream) {
  (void)in_sizes; (void)n_in; (void)out_size; (void)ws_size;
  const float* x     = (const float*)d_in[0];
  const int*   ei    = (const int*)  d_in[1];
  const int*   src   = ei;
  const int*   dst   = ei + N_EDGES;
  const int*   batch = (const int*)  d_in[2];
  const float* emb0  = (const float*)d_in[3];
  const float* emb1  = (const float*)d_in[4];
  const float* emb2  = (const float*)d_in[5];
  const float* emb5  = (const float*)d_in[6];
  const float* Ws[3] = {(const float*)d_in[7],  (const float*)d_in[9],  (const float*)d_in[11]};
  const float* bs[3] = {(const float*)d_in[8],  (const float*)d_in[10], (const float*)d_in[12]};
  const float* gs[3] = {(const float*)d_in[13], (const float*)d_in[15], (const float*)d_in[17]};
  const float* bes[3]= {(const float*)d_in[14], (const float*)d_in[16], (const float*)d_in[18]};
  const float* fcW   = (const float*)d_in[19];
  const float* fcb   = (const float*)d_in[20];
  float* out = (float*)d_out;

  float* ws   = (float*)d_ws;
  float* dinv = ws + OFF_DINV;
  float* hbuf = ws + OFF_H;
  float* hw   = ws + OFF_HW;
  float* agg  = ws + OFF_AGG;
  float* bn   = ws + OFF_BN;
  float* pool = ws + OFF_POOL;
  float* cnt  = ws + OFF_CNT;
  _Float16* apk = (_Float16*)(ws + OFF_HPK);
  _Float16* bpk = (_Float16*)(ws + OFF_WPK);

  const int T = 256;
  // degree -> dinv
  k_fill<<<(N_NODES + T - 1) / T, T, 0, stream>>>(dinv, 1.0f, N_NODES);
  k_deg_scatter<<<(N_EDGES + T - 1) / T, T, 0, stream>>>(dst, dinv);
  k_rsqrt_inplace<<<(N_NODES + T - 1) / T, T, 0, stream>>>(dinv, N_NODES);
  // embedding concat -> h0 (N x 129)
  k_embed<<<(N_NODES * 129 + T - 1) / T, T, 0, stream>>>(x, emb0, emb1, emb2, emb5, hbuf);

  const int gemm_blocks = (NTILES + 7) / 8;           // 8 waves per 256-thread block
  const int nh_blocks   = (N_NODES * HD + T - 1) / T;
  const int edge_blocks = (int)(((long long)N_EDGES * 32 + T - 1) / T);

  for (int layer = 0; layer < 3; ++layer) {
    int K  = (layer == 0) ? 129 : 128;
    int KC = (layer == 0) ? 5 : 4;                    // ceil(K/32)
    int atot = NTILES * KC * 512;                     // packed A halfs
    int btot = KC * 8 * 512;                          // packed B halfs
    k_packA<<<(atot + T - 1) / T, T, 0, stream>>>(hbuf, apk, K, KC, atot);
    k_packB<<<(btot + T - 1) / T, T, 0, stream>>>(Ws[layer], bpk, K, KC, btot);
    if (layer == 0)
      k_gemm_wmma<5><<<gemm_blocks, T, 0, stream>>>(apk, bpk, hw);
    else
      k_gemm_wmma<4><<<gemm_blocks, T, 0, stream>>>(apk, bpk, hw);
    k_selfinit<<<nh_blocks, T, 0, stream>>>(hw, dinv, bs[layer], agg);
    k_edge_scatter<<<edge_blocks, T, 0, stream>>>(src, dst, dinv, hw, agg);
    k_fill<<<1, 256, 0, stream>>>(bn, 0.0f, 256);
    k_bnstats<<<256, HD, 0, stream>>>(agg, bn);
    k_bnapply_relu<<<nh_blocks, T, 0, stream>>>(agg, bn, gs[layer], bes[layer], hbuf); // stride 128 next
  }

  // mean pool + fc
  k_fill<<<(NG * HD + NG + T - 1) / T, T, 0, stream>>>(pool, 0.0f, NG * HD + NG); // pool + cnt contiguous
  k_pool_scatter<<<nh_blocks, T, 0, stream>>>(hbuf, batch, pool);
  k_cnt_scatter<<<(N_NODES + T - 1) / T, T, 0, stream>>>(batch, cnt);
  k_final<<<(NG * NC + T - 1) / T, T, 0, stream>>>(pool, cnt, fcW, fcb, out);
}
